// MultiHeadSelfAttention_90314572300801
// MI455X (gfx1250) — matrix operand order repaired and checked by reading, not compiled, not run on hardware
//
#include <hip/hip_runtime.h>
#include <hip/hip_bf16.h>

// ---------------------------------------------------------------------------
// MI455X (gfx1250) multi-head causal self-attention, f16 WMMA + f32 accum,
// TDM (tensor_load_to_lds) staging of GEMM B tiles, software pipelining.
// ---------------------------------------------------------------------------

typedef _Float16 h16;
typedef __attribute__((ext_vector_type(16))) _Float16 v16h;
typedef __attribute__((ext_vector_type(8)))  _Float16 v8h;
typedef __attribute__((ext_vector_type(4)))  _Float16 v4h;
typedef __attribute__((ext_vector_type(8)))  float    v8f;
typedef __attribute__((ext_vector_type(4)))  unsigned int u32x4;
typedef __attribute__((ext_vector_type(8)))  int          i32x8;
typedef __attribute__((ext_vector_type(4)))  int          i32x4;

#define D_  1024
#define H_  16
#define HD_ 64
#define S_  2048
#define NB_ 2
#define M_  (NB_ * S_)   // 4096 total rows

static __device__ __forceinline__ v16h cat8(v8h a, v8h b) {
  return __builtin_shufflevector(a, b, 0,1,2,3,4,5,6,7,8,9,10,11,12,13,14,15);
}

// A operand of v_wmma_f32_16x16x32_f16 (16 x 32), row-major source, ld in halves.
// Lane L: row M=L%16, K values [16g, 16g+16), g=L/16.
static __device__ __forceinline__ v16h load_a_frag(const h16* base, int ld, int lane) {
  int g = lane >> 4, m = lane & 15;
  const h16* r = base + m * ld;
  v8h lo = *(const v8h*)(r + 16 * g);
  v8h hi = *(const v8h*)(r + 16 * g + 8);
  return cat8(lo, hi);
}

// B operand (math 32x16, K x N); element (k,n) at base[n*ld + k].
// Lane L: column N=L%16, K values [16g, 16g+16)  -> one contiguous v16h.
static __device__ __forceinline__ v16h load_b_frag(const h16* base, int ld, int lane) {
  int g = lane >> 4, n = lane & 15;
  return *(const v16h*)(base + n * ld + 16 * g);
}

// Same, from the 64x64 LDS W tile (ld = 64 halves).
static __device__ __forceinline__ v16h load_b_frag_lds(const h16* base, int lane) {
  int g = lane >> 4, n = lane & 15;
  return *(const v16h*)(base + n * 64 + 16 * g);
}

static __device__ __forceinline__ v8f wmma_f16(v16h a, v16h b, v8f c) {
  return __builtin_amdgcn_wmma_f32_16x16x32_f16(false, a, false, b, (short)0, c,
                                                false, false);
}

// ---------------------------------------------------------------------------
// TDM: DMA a 64-row x 64-half f16 tile (row stride 1024 halves) into LDS.
// D# per CDNA5 ISA.  Tracked by TENSORcnt of the
// issuing wave; EXEC ignored.  6-arg builtin form: groups 0..3 + extra + cpol,
// groups 2/3/extra zero (2-D tensor: tile_dim2 = 0 disables higher dims).
// ---------------------------------------------------------------------------
static __device__ __forceinline__ void tdm_load_tile_64x64(const h16* gsrc,
                                                           unsigned lds_off) {
  unsigned long long ga = (unsigned long long)(uintptr_t)gsrc;
  u32x4 g0;
  g0.x = 1u;                                              // count=1 (valid), user mode
  g0.y = lds_off;                                         // lds_addr (bytes)
  g0.z = (unsigned)(ga & 0xFFFFFFFFu);                    // global_addr[31:0]
  g0.w = (unsigned)((ga >> 32) & 0x1FFFFFFu) | (2u << 30);// addr[56:32] | type=2
  const unsigned td0 = 64, td1 = 64, tl0 = 64, tl1 = 64;  // tensor==tile: no OOB
  const unsigned long long st0 = 1024;                    // dim0 stride (elements)
  i32x8 g1;
  g1[0] = (int)(1u << 16);                                // data_size=1 -> 2 bytes
  g1[1] = (int)((td0 & 0xFFFFu) << 16);                   // tensor_dim0[15:0]
  g1[2] = (int)(((td0 >> 16) & 0xFFFFu) | ((td1 & 0xFFFFu) << 16));
  g1[3] = (int)(((td1 >> 16) & 0xFFFFu) | (tl0 << 16));   // tile_dim0
  g1[4] = (int)(tl1);                                     // tile_dim1; tile_dim2=0
  g1[5] = (int)(st0 & 0xFFFFFFFFull);                     // dim0_stride[31:0]
  g1[6] = (int)((st0 >> 32) & 0xFFFFull);                 // dim0_stride[47:32]; stride1=0
  g1[7] = 0;
  i32x4 gz4 = {0, 0, 0, 0};
  i32x8 gz8 = {0, 0, 0, 0, 0, 0, 0, 0};
  __builtin_amdgcn_tensor_load_to_lds(g0, g1, gz4, gz4, gz8, 0);
}

// ---------------------------------------------------------------------------
// fp32 -> f16 conversion, 4 elements/thread
// ---------------------------------------------------------------------------
__global__ void cvt_f32_f16(const float4* __restrict__ in, v4h* __restrict__ out,
                            int n4) {
  int i = blockIdx.x * blockDim.x + threadIdx.x;
  if (i < n4) {
    float4 v = in[i];
    v4h o;
    o.x = (h16)v.x; o.y = (h16)v.y; o.z = (h16)v.z; o.w = (h16)v.w;
    out[i] = o;
  }
}

// ---------------------------------------------------------------------------
// NT GEMM:  Y[M,1024] = A[M,1024] @ W[1024,1024]^T + bias
// 8 waves/block; wave -> 16x64 tile; block -> 128x64.
// W tile (64 cols x K) staged through LDS by TDM, double-buffered in K-chunks
// of 64; A fragments stream from global with one-chunk register prefetch.
// mode 0: f16 [N,H,S,HD] (Q,K)   mode 1: f16 [N,H,HD,S] (V^T)   mode 2: f32 [M,D]
// ---------------------------------------------------------------------------
__global__ void __launch_bounds__(256) proj_kernel(
    const h16* __restrict__ A, const h16* __restrict__ W,
    const float* __restrict__ bias,
    h16* __restrict__ outh, float* __restrict__ outf, int mode) {
  __shared__ h16 bsh[2][64 * 64];      // 2 x 8KB W-tile buffers

  int tid  = threadIdx.x;
  int wave = tid >> 5, lane = tid & 31;
  int g = lane >> 4, ln = lane & 15;
  int row0 = blockIdx.x * 128 + wave * 16;
  int col0 = blockIdx.y * 64;

  unsigned lds_base = (unsigned)(uintptr_t)&bsh[0][0];
  const h16* Wb = W + (size_t)col0 * D_;     // 64-row W tile for this block

  if (wave == 0) {                           // prologue: chunks 0 and 1 in flight
    tdm_load_tile_64x64(Wb + 0,  lds_base);
    tdm_load_tile_64x64(Wb + 64, lds_base + 8192);
  }

  const h16* Arow = A + (size_t)row0 * D_;
  v16h a0 = load_a_frag(Arow + 0,  D_, lane);
  v16h a1 = load_a_frag(Arow + 32, D_, lane);

  v8f zero = {};
  v8f acc[4] = {zero, zero, zero, zero};

  for (int c = 0; c < 16; ++c) {             // 16 K-chunks of 64
    if (wave == 0) {                         // chunk c complete?
      if (c < 15) __builtin_amdgcn_s_wait_tensorcnt(1);
      else        __builtin_amdgcn_s_wait_tensorcnt(0);
    }
    __syncthreads();

    // register-prefetch A for chunk c+1 (hides global latency behind WMMAs)
    v16h an0 = a0, an1 = a1;
    if (c < 15) {
      an0 = load_a_frag(Arow + (c + 1) * 64,      D_, lane);
      an1 = load_a_frag(Arow + (c + 1) * 64 + 32, D_, lane);
    }

    const h16* bp = &bsh[c & 1][0];
#pragma unroll
    for (int t = 0; t < 4; ++t) {
      v16h b0 = load_b_frag_lds(bp + 16 * t * 64 + 0,  lane);
      acc[t] = wmma_f16(a0, b0, acc[t]);
      v16h b1 = load_b_frag_lds(bp + 16 * t * 64 + 32, lane);
      acc[t] = wmma_f16(a1, b1, acc[t]);
    }

    __syncthreads();                         // all waves done with buf[c&1]
    if (wave == 0 && c + 2 < 16)
      tdm_load_tile_64x64(Wb + (c + 2) * 64, lds_base + (c & 1) * 8192);
    a0 = an0; a1 = an1;
  }

  // C/D layout: lane L, vgpr j -> (row = j + 8*(L/16), col = L%16)
#pragma unroll
  for (int t = 0; t < 4; ++t) {
    int col = col0 + 16 * t + ln;
    float bv = bias[col];
    int hh = col >> 6, dd = col & 63;
#pragma unroll
    for (int j = 0; j < 8; ++j) {
      int r = row0 + j + 8 * g;
      float v = acc[t][j] + bv;
      if (mode == 0) {
        int n = r >> 11, s = r & (S_ - 1);
        outh[(((size_t)(n * H_ + hh)) * S_ + s) * HD_ + dd] = (h16)v;
      } else if (mode == 1) {
        int n = r >> 11, s = r & (S_ - 1);
        outh[(((size_t)(n * H_ + hh)) * HD_ + dd) * S_ + s] = (h16)v;
      } else {
        outf[(size_t)r * D_ + col] = v;
      }
    }
  }
}

// ---------------------------------------------------------------------------
// Causal flash attention.  grid = (N*H, S/128); block = 256 = 8 waves.
// Wave w owns query rows [s0, s0+16).  Q,K: [N,H,S,HD]; Vt: [N,H,HD,S]; O: f16 [N,S,D].
// K fragments are register-prefetched one chunk ahead; V fragments are issued
// before the softmax VALU section so their latency is hidden.
// ---------------------------------------------------------------------------
__global__ void __launch_bounds__(256) attn_kernel(
    const h16* __restrict__ Q, const h16* __restrict__ Km,
    const h16* __restrict__ Vt, h16* __restrict__ O) {
  __shared__ h16 lds_p[8][16 * 32];   // per-wave P staging (C-layout -> A-layout)

  int tid  = threadIdx.x;
  int wave = tid >> 5, lane = tid & 31;
  int g = lane >> 4, ln = lane & 15;
  int nh = blockIdx.x;
  int n = nh >> 4, hh = nh & 15;
  int s0 = blockIdx.y * 128 + wave * 16;

  const h16* Qp = Q  + (size_t)nh * S_ * HD_;
  const h16* Kp = Km + (size_t)nh * S_ * HD_;
  const h16* Vp = Vt + (size_t)nh * HD_ * S_;

  v16h qa0 = load_a_frag(Qp + s0 * HD_ + 0,  HD_, lane);
  v16h qa1 = load_a_frag(Qp + s0 * HD_ + 32, HD_, lane);

  v8f zero = {};
  v8f acc[4] = {zero, zero, zero, zero};
  float mi[8], li[8];
#pragma unroll
  for (int j = 0; j < 8; ++j) { mi[j] = -__builtin_inff(); li[j] = 0.0f; }

  const float isc = 0.125f;   // 1/sqrt(HD)

  // prologue: K fragments for t0 = 0
  v16h kb0 = load_b_frag(Kp + 0,             HD_, lane);
  v16h kb1 = load_b_frag(Kp + 32,            HD_, lane);
  v16h kb2 = load_b_frag(Kp + 16 * HD_,      HD_, lane);
  v16h kb3 = load_b_frag(Kp + 16 * HD_ + 32, HD_, lane);

  for (int t0 = 0; t0 <= s0 + 15; t0 += 32) {
    // ---- scores
    v8f sc0 = zero, sc1 = zero;
    sc0 = wmma_f16(qa0, kb0, sc0);
    sc0 = wmma_f16(qa1, kb1, sc0);
    sc1 = wmma_f16(qa0, kb2, sc1);
    sc1 = wmma_f16(qa1, kb3, sc1);

    // early-issue V fragments for this chunk (consumed after softmax)
    v16h vb0 = load_b_frag(Vp + 0 * S_  + t0, S_, lane);
    v16h vb1 = load_b_frag(Vp + 16 * S_ + t0, S_, lane);
    v16h vb2 = load_b_frag(Vp + 32 * S_ + t0, S_, lane);
    v16h vb3 = load_b_frag(Vp + 48 * S_ + t0, S_, lane);

    // prefetch next chunk's K fragments (wave-uniform branch, EXEC stays full)
    int tn = t0 + 32;
    if (tn <= s0 + 15) {
      kb0 = load_b_frag(Kp + tn * HD_,             HD_, lane);
      kb1 = load_b_frag(Kp + tn * HD_ + 32,        HD_, lane);
      kb2 = load_b_frag(Kp + (tn + 16) * HD_,      HD_, lane);
      kb3 = load_b_frag(Kp + (tn + 16) * HD_ + 32, HD_, lane);
    }

    // ---- causal mask + online softmax (row reductions in 16-lane groups)
#pragma unroll
    for (int j = 0; j < 8; ++j) {
      int srow = s0 + j + 8 * g;
      float x0 = (t0 + ln      <= srow) ? sc0[j] * isc : -__builtin_inff();
      float x1 = (t0 + 16 + ln <= srow) ? sc1[j] * isc : -__builtin_inff();
      float mx = fmaxf(x0, x1);
      mx = fmaxf(mx, __shfl_xor(mx, 1));
      mx = fmaxf(mx, __shfl_xor(mx, 2));
      mx = fmaxf(mx, __shfl_xor(mx, 4));
      mx = fmaxf(mx, __shfl_xor(mx, 8));
      float mnew  = fmaxf(mi[j], mx);          // finite: col 0 never masked
      float p0    = __expf(x0 - mnew);
      float p1    = __expf(x1 - mnew);
      float alpha = __expf(mi[j] - mnew);
      float rs = p0 + p1;
      rs += __shfl_xor(rs, 1);
      rs += __shfl_xor(rs, 2);
      rs += __shfl_xor(rs, 4);
      rs += __shfl_xor(rs, 8);
      li[j] = li[j] * alpha + rs;
      mi[j] = mnew;
#pragma unroll
      for (int t = 0; t < 4; ++t) acc[t][j] *= alpha;
      lds_p[wave][(j + 8 * g) * 32 + ln]      = (h16)p0;
      lds_p[wave][(j + 8 * g) * 32 + 16 + ln] = (h16)p1;
    }

    // per-wave LDS fence (DS in-order within wave; block barrier would hang:
    // waves have divergent trip counts).
    asm volatile("s_wait_dscnt 0" ::: "memory");

    // ---- P (A operand) from LDS, then PV
    const h16* lp = &lds_p[wave][0];
    v8h plo = *(const v8h*)(lp + ln * 32 + 16 * g);
    v8h phi = *(const v8h*)(lp + ln * 32 + 16 * g + 8);
    v16h pa = cat8(plo, phi);
    acc[0] = wmma_f16(pa, vb0, acc[0]);
    acc[1] = wmma_f16(pa, vb1, acc[1]);
    acc[2] = wmma_f16(pa, vb2, acc[2]);
    acc[3] = wmma_f16(pa, vb3, acc[3]);
    asm volatile("" ::: "memory");   // keep next iter's LDS stores after loads
  }

  // ---- epilogue: O16[n*S + s][h*64 + d]
#pragma unroll
  for (int t = 0; t < 4; ++t) {
#pragma unroll
    for (int j = 0; j < 8; ++j) {
      int srow = s0 + j + 8 * g;
      float o = acc[t][j] / li[j];
      O[((size_t)(n * S_ + srow)) * D_ + hh * HD_ + 16 * t + ln] = (h16)o;
    }
  }
}

// ---------------------------------------------------------------------------
extern "C" void kernel_launch(void* const* d_in, const int* in_sizes, int n_in,
                              void* d_out, int out_size, void* d_ws, size_t ws_size,
                              hipStream_t stream) {
  const float* X  = (const float*)d_in[0];
  const float* Wq = (const float*)d_in[1];
  const float* bq = (const float*)d_in[2];
  const float* Wk = (const float*)d_in[3];
  const float* bk = (const float*)d_in[4];
  const float* Wv = (const float*)d_in[5];
  const float* bv = (const float*)d_in[6];
  const float* Wo = (const float*)d_in[7];
  const float* bo = (const float*)d_in[8];
  float* out = (float*)d_out;

  const size_t MK = (size_t)M_ * D_;   // 4,194,304 elements
  const size_t WK = (size_t)D_ * D_;   // 1,048,576 elements

  char* ws = (char*)d_ws;
  h16* X16  = (h16*)ws; ws += MK * 2;
  h16* Wq16 = (h16*)ws; ws += WK * 2;
  h16* Wk16 = (h16*)ws; ws += WK * 2;
  h16* Wv16 = (h16*)ws; ws += WK * 2;
  h16* Wo16 = (h16*)ws; ws += WK * 2;
  h16* Q16  = (h16*)ws; ws += MK * 2;
  h16* K16  = (h16*)ws; ws += MK * 2;
  h16* Vt16 = (h16*)ws; ws += MK * 2;
  h16* O16  = (h16*)ws; ws += MK * 2;   // ~50 MB total

  cvt_f32_f16<<<(int)(MK / 4 / 256), 256, 0, stream>>>((const float4*)X,  (v4h*)X16,  (int)(MK / 4));
  cvt_f32_f16<<<(int)(WK / 4 / 256), 256, 0, stream>>>((const float4*)Wq, (v4h*)Wq16, (int)(WK / 4));
  cvt_f32_f16<<<(int)(WK / 4 / 256), 256, 0, stream>>>((const float4*)Wk, (v4h*)Wk16, (int)(WK / 4));
  cvt_f32_f16<<<(int)(WK / 4 / 256), 256, 0, stream>>>((const float4*)Wv, (v4h*)Wv16, (int)(WK / 4));
  cvt_f32_f16<<<(int)(WK / 4 / 256), 256, 0, stream>>>((const float4*)Wo, (v4h*)Wo16, (int)(WK / 4));

  dim3 pg(M_ / 128, D_ / 64);   // (32, 16)
  proj_kernel<<<pg, 256, 0, stream>>>(X16, Wq16, bq, Q16,  nullptr, 0);
  proj_kernel<<<pg, 256, 0, stream>>>(X16, Wk16, bk, K16,  nullptr, 0);
  proj_kernel<<<pg, 256, 0, stream>>>(X16, Wv16, bv, Vt16, nullptr, 1);

  dim3 ag(NB_ * H_, S_ / 128);  // (32, 16)
  attn_kernel<<<ag, 256, 0, stream>>>(Q16, K16, Vt16, O16);

  proj_kernel<<<pg, 256, 0, stream>>>(O16, Wo16, bo, nullptr, out, 2);
}
